// LFNR_23656679866742
// MI455X (gfx1250) — matrix operand order repaired and checked by reading, not compiled, not run on hardware
//
#include <hip/hip_runtime.h>
#include <math.h>

// ---------------------------------------------------------------------------
// LFNR pipeline for MI455X (gfx1250, wave32).
// All dense GEMMs run on v_wmma_f32_16x16x32_bf16 (bf16 in, f32 accumulate).
// Weights are staged transposed (N x K) so both A and B fragments are
// contiguous 128-bit loads. Each wave computes a 16x32 tile (A reuse).
// Fused epilogues: bias, residual add (in-place into x), GELU -> bf16.
// ---------------------------------------------------------------------------

typedef __attribute__((ext_vector_type(16))) __bf16 v16bf;
typedef __attribute__((ext_vector_type(8)))  __bf16 v8bf;
typedef __attribute__((ext_vector_type(8)))  float  v8f;

#define DEV __device__ __forceinline__

// ---------------- problem constants ----------------
#define BB   2
#define NRR  4096
#define KK   8
#define NN   8
#define HH   224
#define WW   224
#define HW   (HH*WW)          // 50176
#define FDIM 32
#define FEAT 35
#define EMB  32
#define ED   256
#define QDIM 102
#define QPAD 128
#define KDIM 232
#define KPAD 256
#define SEQ  9                // 1 query + 8 keys
#define ROWS (BB*NRR*SEQ)     // 73728 transformer rows
#define BR   (BB*NRR)         // 8192 sequences
#define BRK  (BB*NRR*KK)      // 65536 key rows

// ---------------- input index table (flattened pytree) ----------------
enum {
  IN_GT_O = 0, IN_GT_D, IN_OCC_RGB, IN_GRID, IN_OCC_O, IN_OCC_D, IN_PTS, IN_CAMIDX,
  IN_CONV_W, IN_CONV_B, IN_CAM_EMBED, IN_QUERY_W, IN_QUERY_B, IN_KEY_W, IN_KEY_B,
  IN_LAYER0 = 15,   // +0 ln1w +1 ln1b +2 qkvw +3 qkvb +4 projw +5 projb
                    // +6 ln2w +7 ln2b +8 fc1w +9 fc1b +10 fc2w +11 fc2b
  IN_CORR_W = 39, IN_CORR_B, IN_RGB_W, IN_RGB_B
};

DEV float warp_sum(float v) {
  v += __shfl_xor(v, 1, 32);
  v += __shfl_xor(v, 2, 32);
  v += __shfl_xor(v, 4, 32);
  v += __shfl_xor(v, 8, 32);
  v += __shfl_xor(v, 16, 32);
  return v;
}

DEV v16bf cat8(v8bf lo, v8bf hi) {
  return __builtin_shufflevector(lo, hi, 0,1,2,3,4,5,6,7,8,9,10,11,12,13,14,15);
}

// ---------------------------------------------------------------------------
// bf16 WMMA GEMM: C(MxN f32) = A(MxK bf16, row-major) * Bt^T + bias,
// where Bt is the weight stored TRANSPOSED: Bt[n*ldb + k] (ldb = padded K).
// One wave -> 16(M) x 32(N) via two WMMAs sharing the A fragment.
// 8 waves/block arranged 4(M) x 2(N) -> 64 x 64 block tile. Grids are exact.
// epi 0: C[g*ldc+n] = v            (g = m, or key row-mapping if rowmap==1)
// epi 1: C[m*ldc+n] = v + R[m*ldc+n]   (residual add, R may alias C)
// epi 2: Cb[m*ldc+n] = (bf16) gelu(v)
// ---------------------------------------------------------------------------
__global__ __launch_bounds__(256)
void gemm_bf16_wmma(const __bf16* __restrict__ A, int lda,
                    const __bf16* __restrict__ Bt, int ldb,
                    const float* __restrict__ bias,
                    float* __restrict__ C, __bf16* __restrict__ Cb,
                    const float* __restrict__ R,
                    int ldc, int M, int N, int K, int epi, int rowmap)
{
  const int lane = threadIdx.x & 31;
  const int wv   = threadIdx.x >> 5;
  const int m0 = blockIdx.y * 64 + (wv >> 1) * 16;
  const int n0 = blockIdx.x * 64 + (wv & 1) * 32;
  if (m0 >= M || n0 >= N) return;               // wave-uniform (grid is exact)

  const int l16 = lane & 15;
  const int hi  = lane >> 4;                    // K-half selector

  v8f acc0 = {0.f,0.f,0.f,0.f,0.f,0.f,0.f,0.f};
  v8f acc1 = acc0;

  // A fragment: row (m0+l16); elems 0..15 = K[hi*16 .. +15]
  const __bf16* ap  = A  + (size_t)(m0 + l16) * lda + hi * 16;
  // B fragment (transposed storage): col (n0+l16), 16 contiguous K starting hi*16
  const __bf16* bp0 = Bt + (size_t)(n0 + l16) * ldb + hi * 16;
  const __bf16* bp1 = bp0 + (size_t)16 * ldb;

  for (int k0 = 0; k0 < K; k0 += 32) {
    __builtin_prefetch((const void*)(ap + 64), 0, 3);
    __builtin_prefetch((const void*)(bp0 + 64), 0, 3);
    v16bf a  = cat8(*(const v8bf*)ap, *(const v8bf*)(ap + 8));
    v16bf b0 = *(const v16bf*)bp0;               // 32B aligned
    v16bf b1 = *(const v16bf*)bp1;
    acc0 = __builtin_amdgcn_wmma_f32_16x16x32_bf16(false, a, false, b0,
                                                   (short)0, acc0, false, false);
    acc1 = __builtin_amdgcn_wmma_f32_16x16x32_bf16(false, a, false, b1,
                                                   (short)0, acc1, false, false);
    ap += 32; bp0 += 32; bp1 += 32;
  }

  const int   nC0 = n0 + l16;
  const int   nC1 = nC0 + 16;
  const float bs0 = bias ? bias[nC0] : 0.f;
  const float bs1 = bias ? bias[nC1] : 0.f;
#pragma unroll
  for (int r = 0; r < 8; ++r) {
    const int mC = m0 + r + hi * 8;
    float v0 = acc0[r] + bs0;
    float v1 = acc1[r] + bs1;
    if (epi == 1) {
      size_t i0 = (size_t)mC * ldc + nC0;
      size_t i1 = (size_t)mC * ldc + nC1;
      C[i0] = v0 + R[i0];
      C[i1] = v1 + R[i1];
    } else if (epi == 2) {
      float g0 = 0.5f * v0 * (1.f + tanhf(0.7978845608f * (v0 + 0.044715f * v0 * v0 * v0)));
      float g1 = 0.5f * v1 * (1.f + tanhf(0.7978845608f * (v1 + 0.044715f * v1 * v1 * v1)));
      Cb[(size_t)mC * ldc + nC0] = (__bf16)g0;
      Cb[(size_t)mC * ldc + nC1] = (__bf16)g1;
    } else {
      int g = (rowmap == 1) ? ((mC >> 3) * SEQ + 1 + (mC & 7)) : mC;
      C[(size_t)g * ldc + nC0] = v0;
      C[(size_t)g * ldc + nC1] = v1;
    }
  }
}

// ---------------------------------------------------------------------------
// f32 (Ksrc x N, row-major) -> bf16 transposed (N x Kpad), K zero-padded.
// ---------------------------------------------------------------------------
__global__ void padT_cvt_bf16(const float* __restrict__ src, __bf16* __restrict__ dst,
                              int Ksrc, int Kpad, int N)
{
  int t = blockIdx.x * 256 + threadIdx.x;
  if (t >= Kpad * N) return;
  int n = t / Kpad, kp = t % Kpad;
  dst[t] = (kp < Ksrc) ? (__bf16)src[(size_t)kp * N + n] : (__bf16)0.f;
}

// ---------------------------------------------------------------------------
// 3x3 SAME conv (3->32) + ELU.  One thread per pixel, weights in LDS.
// ---------------------------------------------------------------------------
__global__ __launch_bounds__(256)
void conv3x3_elu(const float* __restrict__ img, const float* __restrict__ cw,
                 const float* __restrict__ cb, float* __restrict__ feat)
{
  __shared__ float sw[FDIM * 27];
  __shared__ float sb[FDIM];
  for (int i = threadIdx.x; i < FDIM * 27; i += 256) sw[i] = cw[i];
  if (threadIdx.x < FDIM) sb[threadIdx.x] = cb[threadIdx.x];
  __syncthreads();

  const int bn = blockIdx.y;
  const int p  = blockIdx.x * 256 + threadIdx.x;      // HW == 196*256 exactly
  const int y = p / WW, x = p % WW;
  const float* base = img + (size_t)bn * 3 * HW;

  float in[27];
#pragma unroll
  for (int ci = 0; ci < 3; ++ci)
#pragma unroll
    for (int ky = 0; ky < 3; ++ky)
#pragma unroll
      for (int kx = 0; kx < 3; ++kx) {
        int iy = y + ky - 1, ix = x + kx - 1;
        bool ok = (iy >= 0) && (iy < HH) && (ix >= 0) && (ix < WW);
        in[ci * 9 + ky * 3 + kx] = ok ? base[(size_t)ci * HW + iy * WW + ix] : 0.f;
      }

  float* fout = feat + (size_t)bn * FDIM * HW + p;
  for (int co = 0; co < FDIM; ++co) {
    float s = sb[co];
    const float* w = &sw[co * 27];
#pragma unroll
    for (int j = 0; j < 27; ++j) s += in[j] * w[j];
    fout[(size_t)co * HW] = (s > 0.f) ? s : (expf(s) - 1.f);
  }
}

// ---------------------------------------------------------------------------
// Bilinear gather of 35 channels (32 conv feat + 3 rgb) per sample.
// ---------------------------------------------------------------------------
__global__ __launch_bounds__(256)
void grid_sample(const float* __restrict__ feat, const float* __restrict__ occ_rgb,
                 const int* __restrict__ camidx, const float* __restrict__ grid,
                 float* __restrict__ proj)
{
  int t = blockIdx.x * 256 + threadIdx.x;            // b*NR*K + r*K + k
  if (t >= BRK) return;
  int b = t / (NRR * KK);
  int cam = camidx[t];
  float gx = grid[(size_t)t * 2 + 0], gy = grid[(size_t)t * 2 + 1];
  float ix = (gx + 1.f) * 0.5f * (WW - 1);
  float iy = (gy + 1.f) * 0.5f * (HH - 1);
  float x0f = floorf(ix), y0f = floorf(iy);
  float wx = ix - x0f, wy = iy - y0f;

  float out[FEAT];
#pragma unroll
  for (int c = 0; c < FEAT; ++c) out[c] = 0.f;

  const float* fbase = feat    + (size_t)(b * NN + cam) * FDIM * HW;
  const float* rbase = occ_rgb + (size_t)(b * NN + cam) * 3 * HW;

#pragma unroll
  for (int corner = 0; corner < 4; ++corner) {
    int dx = corner & 1, dy = corner >> 1;
    float w = (dx ? wx : 1.f - wx) * (dy ? wy : 1.f - wy);
    float xc = x0f + dx, yc = y0f + dy;
    bool valid = (xc >= 0.f) && (xc <= (float)(WW - 1)) &&
                 (yc >= 0.f) && (yc <= (float)(HH - 1));
    if (!valid) continue;
    int xi = (int)xc, yi = (int)yc;
    size_t off = (size_t)yi * WW + xi;
#pragma unroll
    for (int c = 0; c < FDIM; ++c) out[c] += w * fbase[(size_t)c * HW + off];
#pragma unroll
    for (int c = 0; c < 3; ++c) out[FDIM + c] += w * rbase[(size_t)c * HW + off];
  }
  float* po = proj + (size_t)t * FEAT;
#pragma unroll
  for (int c = 0; c < FEAT; ++c) po[c] = out[c];
}

// ---------------------------------------------------------------------------
// Light-field encoding (6 + 48 sin + 48 cos = 102), written as bf16.
// ---------------------------------------------------------------------------
DEV void lf_encode_store(const float o[3], const float din[3], __bf16* dst)
{
  float n = sqrtf(din[0]*din[0] + din[1]*din[1] + din[2]*din[2]) + 1e-8f;
  float d0 = din[0]/n, d1 = din[1]/n, d2 = din[2]/n;
  float m0 = o[1]*d2 - o[2]*d1;
  float m1 = o[2]*d0 - o[0]*d2;
  float m2 = o[0]*d1 - o[1]*d0;
  float x[6] = {d0, d1, d2, m0, m1, m2};
#pragma unroll
  for (int i = 0; i < 6; ++i) dst[i] = (__bf16)x[i];
  for (int s = 0; s < 8; ++s) {
    float sc = (float)(1 << s);
#pragma unroll
    for (int i = 0; i < 6; ++i) {
      float v = x[i] * sc;
      dst[6 + s*6 + i]       = (__bf16)sinf(v);
      dst[6 + 48 + s*6 + i]  = (__bf16)cosf(v);
    }
  }
}

__global__ __launch_bounds__(256)
void build_inq(const float* __restrict__ go, const float* __restrict__ gd,
               __bf16* __restrict__ inq)
{
  int t = blockIdx.x * 256 + threadIdx.x;
  if (t >= BR) return;
  float o[3] = {go[t*3], go[t*3+1], go[t*3+2]};
  float d[3] = {gd[t*3], gd[t*3+1], gd[t*3+2]};
  __bf16* dst = inq + (size_t)t * QPAD;
  lf_encode_store(o, d, dst);
  for (int i = QDIM; i < QPAD; ++i) dst[i] = (__bf16)0.f;
}

__global__ __launch_bounds__(256)
void build_ink(const float* __restrict__ oo, const float* __restrict__ od,
               const float* __restrict__ pts, const int* __restrict__ camidx,
               const float* __restrict__ cam_embed, const float* __restrict__ proj,
               __bf16* __restrict__ ink)
{
  int t = blockIdx.x * 256 + threadIdx.x;            // b*NR*K + r*K + k
  if (t >= BRK) return;
  int br = t >> 3;                                   // b*NR + r
  float o[3] = {oo[t*3], oo[t*3+1], oo[t*3+2]};
  float d[3] = {od[t*3], od[t*3+1], od[t*3+2]};
  __bf16* dst = ink + (size_t)t * KPAD;
  lf_encode_store(o, d, dst);                        // [0,102)
  // positional encoding of pts_3d (3 + 30 sin + 30 cos = 63) at [102,165)
  float p0 = pts[br*3], p1 = pts[br*3+1], p2 = pts[br*3+2];
  __bf16* w = dst + QDIM;
  w[0] = (__bf16)p0; w[1] = (__bf16)p1; w[2] = (__bf16)p2;
  for (int s = 0; s < 10; ++s) {
    float sc = (float)(1 << s);
    float v0 = p0*sc, v1 = p1*sc, v2 = p2*sc;
    w[3 + s*3 + 0]  = (__bf16)sinf(v0);
    w[3 + s*3 + 1]  = (__bf16)sinf(v1);
    w[3 + s*3 + 2]  = (__bf16)sinf(v2);
    w[33 + s*3 + 0] = (__bf16)cosf(v0);
    w[33 + s*3 + 1] = (__bf16)cosf(v1);
    w[33 + s*3 + 2] = (__bf16)cosf(v2);
  }
  const float* pr = proj + (size_t)t * FEAT;         // projected at [165,200)
#pragma unroll
  for (int c = 0; c < FEAT; ++c) dst[165 + c] = (__bf16)pr[c];
  const float* em = cam_embed + camidx[t] * EMB;     // cam embed at [200,232)
#pragma unroll
  for (int c = 0; c < EMB; ++c) dst[200 + c] = (__bf16)em[c];
  for (int c = KDIM; c < KPAD; ++c) dst[c] = (__bf16)0.f;
}

// ---------------------------------------------------------------------------
// LayerNorm over 256 cols, one wave per row, bf16 output.
// ---------------------------------------------------------------------------
__global__ __launch_bounds__(256)
void layernorm_bf16(const float* __restrict__ x, const float* __restrict__ w,
                    const float* __restrict__ b, __bf16* __restrict__ h, int rows)
{
  int row = blockIdx.x * 8 + (threadIdx.x >> 5);
  if (row >= rows) return;
  int lane = threadIdx.x & 31;
  const float* xr = x + (size_t)row * ED;
  float v[8], s = 0.f, s2 = 0.f;
#pragma unroll
  for (int j = 0; j < 8; ++j) { v[j] = xr[lane + 32*j]; s += v[j]; s2 += v[j]*v[j]; }
  s = warp_sum(s); s2 = warp_sum(s2);
  float mu  = s  * (1.f/ED);
  float var = s2 * (1.f/ED) - mu*mu;
  float inv = rsqrtf(var + 1e-5f);
  __bf16* hr = h + (size_t)row * ED;
#pragma unroll
  for (int j = 0; j < 8; ++j) {
    int c = lane + 32*j;
    hr[c] = (__bf16)((v[j] - mu) * inv * w[c] + b[c]);
  }
}

// ---------------------------------------------------------------------------
// MHA attention core, S=9, 4 heads x dh=64. Thread = (seq, head, s-pos).
// Reads packed qkv f32 (rows x 768), writes o in bf16 (rows x 256).
// ---------------------------------------------------------------------------
__global__ __launch_bounds__(256)
void attention(const float* __restrict__ qkv, __bf16* __restrict__ o)
{
  int t = blockIdx.x * 256 + threadIdx.x;            // seq*36 + head*9 + s
  if (t >= BR * 4 * SEQ) return;
  int s   = t % SEQ;
  int hd  = (t / SEQ) & 3;
  int seq = t / (SEQ * 4);
  const float* base = qkv + (size_t)seq * SEQ * 768;
  const float* qr = base + (size_t)s * 768 + hd * 64;
  float q[64];
#pragma unroll
  for (int d = 0; d < 64; ++d) q[d] = qr[d];

  float sc[SEQ]; float mx = -1e30f;
  for (int tt = 0; tt < SEQ; ++tt) {
    const float* kr = base + (size_t)tt * 768 + ED + hd * 64;
    float a = 0.f;
#pragma unroll
    for (int d = 0; d < 64; ++d) a += q[d] * kr[d];
    a *= 0.125f;                                     // 1/sqrt(64)
    sc[tt] = a; mx = fmaxf(mx, a);
  }
  float den = 0.f;
  for (int tt = 0; tt < SEQ; ++tt) { sc[tt] = expf(sc[tt] - mx); den += sc[tt]; }
  float inv = 1.f / den;

  float out[64];
#pragma unroll
  for (int d = 0; d < 64; ++d) out[d] = 0.f;
  for (int tt = 0; tt < SEQ; ++tt) {
    const float* vr = base + (size_t)tt * 768 + 2*ED + hd * 64;
    float wgt = sc[tt] * inv;
#pragma unroll
    for (int d = 0; d < 64; ++d) out[d] += wgt * vr[d];
  }
  __bf16* orow = o + (size_t)(seq * SEQ + s) * ED + hd * 64;
#pragma unroll
  for (int d = 0; d < 64; ++d) orow[d] = (__bf16)out[d];
}

// ---------------------------------------------------------------------------
// Final head: corr logits -> softmax over K, combine, rgb sigmoid, overlap.
// One wave per (b,r) sequence; lanes split the 256 channels.
// ---------------------------------------------------------------------------
__global__ __launch_bounds__(256)
void final_head(const float* __restrict__ x, const float* __restrict__ proj,
                const float* __restrict__ cw, const float* __restrict__ cb,
                const float* __restrict__ rw, const float* __restrict__ rb,
                float* __restrict__ out)
{
  int row = blockIdx.x * 8 + (threadIdx.x >> 5);     // b*NR + r
  if (row >= BR) return;
  int lane = threadIdx.x & 31;
  const float* xq = x + (size_t)row * SEQ * ED;

  float lq = 0.f;
#pragma unroll
  for (int j = 0; j < 8; ++j) { int c = lane*8 + j; lq += xq[c] * cw[c]; }
  lq = warp_sum(lq);

  float logit[KK]; float mx = -1e30f;
  for (int k = 0; k < KK; ++k) {
    const float* xk = xq + (size_t)(1 + k) * ED;
    float p = 0.f;
#pragma unroll
    for (int j = 0; j < 8; ++j) { int c = lane*8 + j; p += xk[c] * cw[ED + c]; }
    p = warp_sum(p);
    logit[k] = lq + p + cb[0];
    mx = fmaxf(mx, logit[k]);
  }
  float den = 0.f, att[KK];
  for (int k = 0; k < KK; ++k) { att[k] = expf(logit[k] - mx); den += att[k]; }
  float inv = 1.f / den;
  for (int k = 0; k < KK; ++k) att[k] *= inv;

  float comb[8];
#pragma unroll
  for (int j = 0; j < 8; ++j) comb[j] = 0.f;
  for (int k = 0; k < KK; ++k) {
    const float* xk = xq + (size_t)(1 + k) * ED;
    float a = att[k];
#pragma unroll
    for (int j = 0; j < 8; ++j) comb[j] += a * xk[lane*8 + j];
  }
  for (int j3 = 0; j3 < 3; ++j3) {
    float p = 0.f;
#pragma unroll
    for (int j = 0; j < 8; ++j) { int c = lane*8 + j; p += comb[j] * rw[c*3 + j3]; }
    p = warp_sum(p);
    if (lane == 0)
      out[(size_t)row * 3 + j3] = 1.f / (1.f + expf(-(p + rb[j3])));
  }
  if (lane == 0) {
    float ov[3] = {0.f, 0.f, 0.f};
    for (int k = 0; k < KK; ++k) {
      const float* pr = proj + ((size_t)row * KK + k) * FEAT + FDIM;  // last 3 ch
      for (int j = 0; j < 3; ++j) ov[j] += att[k] * pr[j];
    }
    for (int j = 0; j < 3; ++j)
      out[(size_t)BR * 3 + (size_t)row * 3 + j] = ov[j];
  }
}

// ---------------------------------------------------------------------------
// Host launcher
// ---------------------------------------------------------------------------
extern "C" void kernel_launch(void* const* d_in, const int* in_sizes, int n_in,
                              void* d_out, int out_size, void* d_ws, size_t ws_size,
                              hipStream_t stream)
{
  (void)in_sizes; (void)n_in; (void)out_size; (void)ws_size;
  char* W = (char*)d_ws;

  // ---- workspace layout (lifetime-overlapped) ----
  const size_t SZ_UA   = (size_t)ROWS * 768 * 4;       // feat(103MB) U qkv(226MB)
  const size_t OFF_UB  = SZ_UA;
  const size_t SZ_UB   = (size_t)ROWS * 1024 * 2;      // g1 U (inq + ink)
  const size_t OFF_X   = OFF_UB + SZ_UB;
  const size_t SZ_X    = (size_t)ROWS * ED * 4;
  const size_t OFF_H   = OFF_X + SZ_X;
  const size_t SZ_H    = (size_t)ROWS * ED * 2;
  const size_t OFF_O   = OFF_H + SZ_H;
  const size_t OFF_PRJ = OFF_O + SZ_H;
  const size_t SZ_PRJ  = (size_t)BRK * FEAT * 4;
  size_t woff = OFF_PRJ + SZ_PRJ;

  float*  feat   = (float*)(W + 0);
  float*  qkvbuf = (float*)(W + 0);
  __bf16* inq    = (__bf16*)(W + OFF_UB);
  __bf16* ink    = (__bf16*)(W + OFF_UB + (size_t)BR * QPAD * 2);
  __bf16* g1     = (__bf16*)(W + OFF_UB);
  float*  x      = (float*)(W + OFF_X);
  __bf16* h      = (__bf16*)(W + OFF_H);
  __bf16* obuf   = (__bf16*)(W + OFF_O);
  float*  proj   = (float*)(W + OFF_PRJ);

  // bf16 transposed weight staging (N x Kpad each)
  __bf16* wq = (__bf16*)(W + woff); woff += (size_t)QPAD * ED * 2;
  __bf16* wk = (__bf16*)(W + woff); woff += (size_t)KPAD * ED * 2;
  __bf16* wqkv[2]; __bf16* wproj[2]; __bf16* wfc1[2]; __bf16* wfc2[2];
  for (int l = 0; l < 2; ++l) {
    wqkv[l] = (__bf16*)(W + woff); woff += (size_t)ED * 768 * 2;
    wproj[l] = (__bf16*)(W + woff); woff += (size_t)ED * ED * 2;
    wfc1[l] = (__bf16*)(W + woff); woff += (size_t)ED * 1024 * 2;
    wfc2[l] = (__bf16*)(W + woff); woff += (size_t)1024 * ED * 2;
  }

  // ---- 0) weight conversion f32 -> bf16, transposed to (N, Kpad) ----
  padT_cvt_bf16<<<(QPAD*ED + 255)/256, 256, 0, stream>>>(
      (const float*)d_in[IN_QUERY_W], wq, QDIM, QPAD, ED);
  padT_cvt_bf16<<<(KPAD*ED + 255)/256, 256, 0, stream>>>(
      (const float*)d_in[IN_KEY_W], wk, KDIM, KPAD, ED);
  for (int l = 0; l < 2; ++l) {
    int base = IN_LAYER0 + l * 12;
    padT_cvt_bf16<<<(ED*768 + 255)/256, 256, 0, stream>>>(
        (const float*)d_in[base + 2], wqkv[l], ED, ED, 768);
    padT_cvt_bf16<<<(ED*ED + 255)/256, 256, 0, stream>>>(
        (const float*)d_in[base + 4], wproj[l], ED, ED, ED);
    padT_cvt_bf16<<<(ED*1024 + 255)/256, 256, 0, stream>>>(
        (const float*)d_in[base + 8], wfc1[l], ED, ED, 1024);
    padT_cvt_bf16<<<(1024*ED + 255)/256, 256, 0, stream>>>(
        (const float*)d_in[base + 10], wfc2[l], 1024, 1024, ED);
  }

  // ---- 1) conv3x3 + ELU over 16 images ----
  conv3x3_elu<<<dim3(HW/256, BB*NN), 256, 0, stream>>>(
      (const float*)d_in[IN_OCC_RGB], (const float*)d_in[IN_CONV_W],
      (const float*)d_in[IN_CONV_B], feat);

  // ---- 2) bilinear gather -> projected (B*NR*K, 35) ----
  grid_sample<<<BRK/256, 256, 0, stream>>>(
      feat, (const float*)d_in[IN_OCC_RGB], (const int*)d_in[IN_CAMIDX],
      (const float*)d_in[IN_GRID], proj);

  // ---- 3) encodings -> padded bf16 GEMM inputs ----
  build_inq<<<BR/256, 256, 0, stream>>>(
      (const float*)d_in[IN_GT_O], (const float*)d_in[IN_GT_D], inq);
  build_ink<<<BRK/256, 256, 0, stream>>>(
      (const float*)d_in[IN_OCC_O], (const float*)d_in[IN_OCC_D],
      (const float*)d_in[IN_PTS], (const int*)d_in[IN_CAMIDX],
      (const float*)d_in[IN_CAM_EMBED], proj, ink);

  // ---- 4) query / key linears write straight into x (8192 x 9 x 256) ----
  gemm_bf16_wmma<<<dim3(ED/64, BR/64), 256, 0, stream>>>(
      inq, QPAD, wq, QPAD, (const float*)d_in[IN_QUERY_B],
      x, nullptr, nullptr, SEQ*ED, BR, ED, QPAD, /*epi=*/0, /*rowmap=*/0);
  gemm_bf16_wmma<<<dim3(ED/64, BRK/64), 256, 0, stream>>>(
      ink, KPAD, wk, KPAD, (const float*)d_in[IN_KEY_B],
      x, nullptr, nullptr, ED, BRK, ED, KPAD, /*epi=*/0, /*rowmap=*/1);

  // ---- 5) transformer (2 layers) ----
  for (int l = 0; l < 2; ++l) {
    int base = IN_LAYER0 + l * 12;
    // h = LN1(x)
    layernorm_bf16<<<ROWS/8, 256, 0, stream>>>(
        x, (const float*)d_in[base + 0], (const float*)d_in[base + 1], h, ROWS);
    // qkv = h @ Wqkv + b  (f32, overlays feat region)
    gemm_bf16_wmma<<<dim3(768/64, ROWS/64), 256, 0, stream>>>(
        h, ED, wqkv[l], ED, (const float*)d_in[base + 3],
        qkvbuf, nullptr, nullptr, 768, ROWS, 768, ED, 0, 0);
    // attention -> obuf (bf16)
    attention<<<(BR*4*SEQ)/256, 256, 0, stream>>>(qkvbuf, obuf);
    // x += obuf @ Wproj + b
    gemm_bf16_wmma<<<dim3(ED/64, ROWS/64), 256, 0, stream>>>(
        obuf, ED, wproj[l], ED, (const float*)d_in[base + 5],
        x, nullptr, x, ED, ROWS, ED, ED, /*epi=*/1, 0);
    // h = LN2(x)
    layernorm_bf16<<<ROWS/8, 256, 0, stream>>>(
        x, (const float*)d_in[base + 6], (const float*)d_in[base + 7], h, ROWS);
    // g1 = gelu(h @ Wfc1 + b)  (bf16, fused epilogue)
    gemm_bf16_wmma<<<dim3(1024/64, ROWS/64), 256, 0, stream>>>(
        h, ED, wfc1[l], ED, (const float*)d_in[base + 9],
        nullptr, g1, nullptr, 1024, ROWS, 1024, ED, /*epi=*/2, 0);
    // x += g1 @ Wfc2 + b
    gemm_bf16_wmma<<<dim3(ED/64, ROWS/64), 256, 0, stream>>>(
        g1, 1024, wfc2[l], 1024, (const float*)d_in[base + 11],
        x, nullptr, x, ED, ROWS, ED, 1024, /*epi=*/1, 0);
  }

  // ---- 6) final head: corr softmax, combine, rgb + rgb_overlap ----
  final_head<<<BR/8, 256, 0, stream>>>(
      x, proj,
      (const float*)d_in[IN_CORR_W], (const float*)d_in[IN_CORR_B],
      (const float*)d_in[IN_RGB_W], (const float*)d_in[IN_RGB_B],
      (float*)d_out);
}
